// Hierarchical_Message_Passing_Structure_base_41970420417491
// MI455X (gfx1250) — matrix operand order repaired and checked by reading, not compiled, not run on hardware
//
#include <hip/hip_runtime.h>
#include <hip/hip_bf16.h>
#include <stdint.h>

typedef __attribute__((ext_vector_type(16))) __bf16 v16bf;
typedef __attribute__((ext_vector_type(8)))  float  v8f;

#define D        256     // feature size
#define KDIM     512     // 2*D (gate input)
#define GW       128     // gate width
#define NT_TILES 8       // GW / 16
#define KSTEPS   16      // KDIM / 32
#define WAVES    4
#define BLOCK    (WAVES * 32)
#define FRAG_ELEMS (NT_TILES * KSTEPS * 32 * 16)   // 65536 bf16 = 128 KB

__device__ __forceinline__ __bf16 relu_bf(float f) {
    return (__bf16)fmaxf(f, 0.0f);
}

// Fast sigmoid: v_exp_f32 + v_add + v_rcp_f32 (avoids the precise-division
// v_div_scale/v_div_fmas expansion; the result feeds a 128-way mean, so the
// ~1 ulp rcp error is irrelevant).
__device__ __forceinline__ float sigmoidf_fast(float x) {
    return __builtin_amdgcn_rcpf(1.0f + __expf(-x));
}

// ---------------------------------------------------------------------------
// Kernel 0: zero output accumulators and per-node counts
// ---------------------------------------------------------------------------
__global__ void k_init(float* __restrict__ out, float* __restrict__ counts,
                       int n_out, int n_counts) {
    int i = blockIdx.x * blockDim.x + threadIdx.x;
    int stride = gridDim.x * blockDim.x;
    for (int j = i; j < n_out; j += stride)    out[j] = 0.0f;
    for (int j = i; j < n_counts; j += stride) counts[j] = 0.0f;
}

// ---------------------------------------------------------------------------
// Kernel 1: pack w (fp32 [128][512], row-major) into bf16 WMMA B-fragments.
// Fragment (nt, ks) covers N = nt*16..+15, K = ks*32..+31 of w^T (K x N).
// Per ISA 7.12.2 B layout (wave32): lane -> N = nt*16 + (L&15);
// lanes 0-15: element i -> K = i (i<8), 16 + (i-8) (i>=8); lanes 16-31:
// element i -> K = 8 + i (i<8), 24 + (i-8) (i>=8).
// Flat layout: wfrag[(((nt*KSTEPS + ks)*32 + L)*16) + i]
// ---------------------------------------------------------------------------
__global__ void k_pack_w(const float* __restrict__ w, __bf16* __restrict__ wfrag) {
    int tid = blockIdx.x * blockDim.x + threadIdx.x;
    if (tid >= NT_TILES * KSTEPS * 32) return;
    int L  = tid & 31;
    int f  = tid >> 5;           // f = nt*KSTEPS + ks
    int ks = f & (KSTEPS - 1);
    int nt = f / KSTEPS;
    int n  = nt * 16 + (L & 15);
    int kg = ks * 32 + ((L >= 16) ? 8 : 0);
#pragma unroll
    for (int i = 0; i < 16; ++i) {
        wfrag[tid * 16 + i] = (__bf16)w[n * KDIM + (kg + i + ((i >= 8) ? 8 : 0))];
    }
}

// ---------------------------------------------------------------------------
// Kernel 2: main message-passing kernel.
// Each wave handles a tile of 16 edges: A (16x512, relu'd concat feats, bf16)
// times B (512x128 = w^T, bf16 from LDS) via 8x16 = 128 v_wmma_f32_16x16x32_bf16,
// then sigmoid + mean -> per-edge gate, then gated fp32 atomic scatter.
// ---------------------------------------------------------------------------
__global__ void __launch_bounds__(BLOCK)
k_mpn(const float* __restrict__ tgtf, const float* __restrict__ srcf,
      const long long* __restrict__ etgt, const long long* __restrict__ esrc,
      const float* __restrict__ bias, const __bf16* __restrict__ wfrag,
      float* __restrict__ out, float* __restrict__ counts,
      int E, int nTiles) {
    extern __shared__ __bf16 wlds[];   // 65536 bf16 = 128 KB

    // Cooperative stage of packed W fragments into LDS (uint4 = 16B chunks).
    {
        const uint4* s4 = reinterpret_cast<const uint4*>(wfrag);
        uint4* d4 = reinterpret_cast<uint4*>(wlds);
        for (int i = threadIdx.x; i < FRAG_ELEMS / 8; i += blockDim.x)
            d4[i] = s4[i];
    }
    __syncthreads();

    const int lane = threadIdx.x & 31;
    const int wave = threadIdx.x >> 5;
    const int m    = lane & 15;          // edge-row within tile this lane owns
    const bool hi  = (lane >= 16);

    // Per-lane bias for gate column n = nt*16 + m (C/D layout: N = lane&15).
    float bv[NT_TILES];
#pragma unroll
    for (int nt = 0; nt < NT_TILES; ++nt) bv[nt] = bias[nt * 16 + m];

    const v16bf* bfrags = reinterpret_cast<const v16bf*>(wlds);
    const v8f vzero = {0.f, 0.f, 0.f, 0.f, 0.f, 0.f, 0.f, 0.f};
    const int tileStride = gridDim.x * WAVES;

    for (int tile = blockIdx.x * WAVES + wave; tile < nTiles; tile += tileStride) {
        const int e0 = tile * 16;
        int e = e0 + m;
        if (e >= E) e = E - 1;                 // clamp (scatter guarded below)
        const int ti = (int)etgt[e];
        const int si = (int)esrc[e];
        const float* trow = tgtf + (long)ti * D;
        const float* srow = srcf + (long)si * D;

        // Prefetch next tile's edge indices (global_prefetch_b8).
        {
            int en = e0 + tileStride * 16 + m;
            if (en < E) {
                __builtin_prefetch(etgt + en, 0, 0);
                __builtin_prefetch(esrc + en, 0, 0);
            }
        }

        v8f acc[NT_TILES];
#pragma unroll
        for (int nt = 0; nt < NT_TILES; ++nt) acc[nt] = vzero;

        // K loop: ks 0..7 -> relu(tgt row), ks 8..15 -> relu(src row).
        for (int ks = 0; ks < KSTEPS; ++ks) {
            const float* row = (ks < 8) ? trow : srow;
            const int kk = (ks & 7) * 32 + (hi ? 8 : 0);
            // A 16-bit layout: elements 0..7 -> K=kk..kk+7, 8..15 -> kk+16..kk+23
            const float4* p = reinterpret_cast<const float4*>(row + kk);
            float4 x0 = p[0], x1 = p[1], x2 = p[4], x3 = p[5];
            v16bf a;
            a[0]  = relu_bf(x0.x); a[1]  = relu_bf(x0.y);
            a[2]  = relu_bf(x0.z); a[3]  = relu_bf(x0.w);
            a[4]  = relu_bf(x1.x); a[5]  = relu_bf(x1.y);
            a[6]  = relu_bf(x1.z); a[7]  = relu_bf(x1.w);
            a[8]  = relu_bf(x2.x); a[9]  = relu_bf(x2.y);
            a[10] = relu_bf(x2.z); a[11] = relu_bf(x2.w);
            a[12] = relu_bf(x3.x); a[13] = relu_bf(x3.y);
            a[14] = relu_bf(x3.z); a[15] = relu_bf(x3.w);

#pragma unroll
            for (int nt = 0; nt < NT_TILES; ++nt) {
                v16bf bf = bfrags[(nt * KSTEPS + ks) * 32 + lane];
                acc[nt] = __builtin_amdgcn_wmma_f32_16x16x32_bf16(
                    false, a, false, bf, (short)0, acc[nt], false, false);
            }
        }

        // Gate: mean over 128 sigmoid outputs per edge.
        // C/D layout: acc[nt][r] is (M = r + (hi?8:0), N = nt*16 + m).
        float g[8];
#pragma unroll
        for (int r = 0; r < 8; ++r) {
            float s = 0.0f;
#pragma unroll
            for (int nt = 0; nt < NT_TILES; ++nt)
                s += sigmoidf_fast(acc[nt][r] + bv[nt]);
            // Reduce across the 16 lanes that share this M (xor stays in half).
            s += __shfl_xor(s, 1, 32);
            s += __shfl_xor(s, 2, 32);
            s += __shfl_xor(s, 4, 32);
            s += __shfl_xor(s, 8, 32);
            g[r] = s * (1.0f / (float)GW);
        }
        // Lanes 0-15 now hold gates for edges M=0..7 (g[0..7]),
        // lanes 16-31 for edges M=8..15.

        // Scatter: transferred = src_f * gate, fp32 atomics into out[tgt].
        for (int me = 0; me < 16; ++me) {
            int ee = e0 + me;
            if (ee >= E) break;
            float gm = __shfl(g[me & 7], (me < 8) ? 0 : 16, 32);
            int tim  = __shfl(ti, me, 32);
            int sim  = __shfl(si, me, 32);
            const float* sr = srcf + (long)sim * D;
            float* orow = out + (long)tim * D;
            int c = lane * 8;                 // 32 lanes x 8 cols = 256
#pragma unroll
            for (int j = 0; j < 8; ++j) {
                float v = sr[c + j] * gm;
                __hip_atomic_fetch_add(orow + c + j, v,
                                       __ATOMIC_RELAXED, __HIP_MEMORY_SCOPE_AGENT);
            }
        }
        // One count per edge (lanes 0..15 each own one edge of the tile).
        if (!hi && (e0 + m) < E) {
            __hip_atomic_fetch_add(counts + ti, 1.0f,
                                   __ATOMIC_RELAXED, __HIP_MEMORY_SCOPE_AGENT);
        }
    }
}

// ---------------------------------------------------------------------------
// Kernel 3: out = sums / counts where counts > 0 (rows with 0 stay 0).
// Precise division here (directly feeds the output).
// ---------------------------------------------------------------------------
__global__ void k_finalize(float* __restrict__ out, const float* __restrict__ counts,
                           int total) {
    int i = blockIdx.x * blockDim.x + threadIdx.x;
    int stride = gridDim.x * blockDim.x;
    for (int j = i; j < total; j += stride) {
        float c = counts[j >> 8];            // D == 256
        out[j] = (c > 0.0f) ? out[j] / c : 0.0f;
    }
}

// ---------------------------------------------------------------------------
extern "C" void kernel_launch(void* const* d_in, const int* in_sizes, int n_in,
                              void* d_out, int out_size, void* d_ws, size_t ws_size,
                              hipStream_t stream) {
    const float*     tgtf = (const float*)d_in[0];
    const float*     srcf = (const float*)d_in[1];
    const long long* etgt = (const long long*)d_in[2];   // int64 in reference
    const long long* esrc = (const long long*)d_in[3];   // int64 in reference
    const float*     w    = (const float*)d_in[4];
    const float*     b    = (const float*)d_in[5];
    float* out = (float*)d_out;

    const int E  = in_sizes[2];
    const int nT = in_sizes[0] / D;

    // Workspace: [counts: nT floats][packed bf16 W fragments: 128 KB]
    float*  counts = (float*)d_ws;
    size_t  coff   = ((size_t)nT * sizeof(float) + 255) & ~(size_t)255;
    __bf16* wfrag  = (__bf16*)((char*)d_ws + coff);

    k_init<<<512, 256, 0, stream>>>(out, counts, nT * D, nT);
    k_pack_w<<<(NT_TILES * KSTEPS * 32 + 255) / 256, 256, 0, stream>>>(w, wfrag);

    int nTiles = (E + 15) / 16;
    int grid = 1024;
    if (grid * WAVES > nTiles) grid = (nTiles + WAVES - 1) / WAVES;
    k_mpn<<<grid, BLOCK, FRAG_ELEMS * sizeof(__bf16), stream>>>(
        tgtf, srcf, etgt, esrc, b, wfrag, out, counts, E, nTiles);

    k_finalize<<<1024, 256, 0, stream>>>(out, counts, nT * D);
}
